// MultiHeadAttention_64622077935948
// MI455X (gfx1250) — matrix operand order repaired and checked by reading, not compiled, not run on hardware
//
#include <hip/hip_runtime.h>

typedef _Float16 f16;
typedef _Float16 v16h __attribute__((ext_vector_type(16)));
typedef float    v8f  __attribute__((ext_vector_type(8)));
typedef unsigned int u32x4 __attribute__((ext_vector_type(4)));
typedef int          i32x4 __attribute__((ext_vector_type(4)));
typedef int          i32x8 __attribute__((ext_vector_type(8)));

union AFrag { v16h v; f16   e[16]; uint4 q[2]; };
union CFrag { v8f  v; float e[8];  };

#define D_MODEL 1024
#define N_HEADS 16
#define DEPTH   64
#define BATCH   2
#define SEQ     2048
#define M_TOT   (BATCH * SEQ)   // 4096

#if defined(__HIP_DEVICE_COMPILE__) && \
    __has_builtin(__builtin_amdgcn_tensor_load_to_lds) && \
    __has_builtin(__builtin_amdgcn_s_wait_tensorcnt)
#define USE_TDM 1
#else
#define USE_TDM 0
#endif

// width-16 XOR shuffle: v_permlane16_b32 (pure VALU, co-executes with WMMA)
// instead of ds_bpermute round-trips.
#if defined(__HIP_DEVICE_COMPILE__) && __has_builtin(__builtin_amdgcn_permlane16)
#define XSHF16(x, m, s0, s1)                                                  \
    __int_as_float(__builtin_amdgcn_permlane16(                               \
        __float_as_int(x), __float_as_int(x), (int)(s0), (int)(s1),           \
        false, false))
#else
#define XSHF16(x, m, s0, s1) __shfl_xor((x), (m), 16)
#endif
// XOR selector tables (nibble = source lane within 16-group)
#define SEL1_LO 0x67452301u
#define SEL1_HI 0xEFCDAB89u
#define SEL2_LO 0x54761032u
#define SEL2_HI 0xDCFE98BAu
#define SEL4_LO 0x32107654u
#define SEL4_HI 0xBA98FEDCu
#define SEL8_LO 0xFEDCBA98u
#define SEL8_HI 0x76543210u

#if USE_TDM
// Issue one TDM 2D tile load: 32 rows x 64 f16 cols, row stride D_MODEL elems.
// D# per CDNA5 ISA ch.8: group0 = {flags, lds_addr, global_addr lo, hi|type=2},
// group1 = {data_size=2B, tensor_dim0=64, tensor_dim1=SEQ, tile 64x32, stride0=1024}.
// This toolchain exposes the 6-arg builtin: (g0, g1, g2, g3, g4, cpol).
__device__ __forceinline__ void tdm_load_tile_32x64(unsigned long long gaddr,
                                                    unsigned ldsaddr)
{
    u32x4 g0 = { 1u,                       // count=1, no gather
                 ldsaddr,                  // LDS byte address
                 (unsigned)gaddr,          // global_addr[31:0]
                 (unsigned)((gaddr >> 32) & 0x01FFFFFFu) | (2u << 30) }; // [56:32]|type=2
    i32x8 g1 = { (int)(1u << 16),          // workgroup_mask=0, data_size=1 (2 bytes)
                 (int)(64u << 16),         // atomic_barrier=0 | tensor_dim0.lo16 = 64
                 (int)((unsigned)SEQ << 16), // tensor_dim0.hi16=0 | tensor_dim1.lo16
                 (int)(64u << 16),         // tensor_dim1.hi16=0 | tile_dim0 = 64
                 (int)32,                  // tile_dim1 = 32 | tile_dim2 = 0
                 (int)D_MODEL,             // tensor_dim0_stride[31:0] = 1024
                 0, 0 };                   // stride0.hi16 / stride1 = 0
    i32x4 g2 = { 0, 0, 0, 0 };
    i32x4 g3 = { 0, 0, 0, 0 };
    i32x8 g4 = { 0, 0, 0, 0, 0, 0, 0, 0 };
    __builtin_amdgcn_tensor_load_to_lds(g0, g1, g2, g3, g4, 0);
}
#endif

// ---------------------------------------------------------------------------
// Fused convert(f32->f16) + GEMM + bias on the WMMA f16 path.
//   out = A[4096x1024](f32) * W[1024x1024](f32) + bias
// Block: 256 threads (8 waves), tile 128x64. Waves 4x2, each computes a 32x32
// macro-tile = 2x2 WMMA 16x16x32 per K-chunk with fragment reuse.
// B is staged TRANSPOSED in LDS so B fragments are contiguous b128 loads.
// ---------------------------------------------------------------------------
template <bool F32OUT>
__global__ __launch_bounds__(256)
void gemm_wmma(const float* __restrict__ A, const float* __restrict__ W,
               const float* __restrict__ bias, f16* __restrict__ outH,
               float* __restrict__ outF)
{
    __shared__ f16 As [128 * 32];   // [row][k]
    __shared__ f16 Bst[64  * 32];   // [col][k]  (transposed)

    const int tid  = threadIdx.x;
    const int lane = tid & 31;
    const int wave = tid >> 5;
    const int wm   = wave >> 1;            // 0..3 : 32-row group
    const int wn   = wave & 1;             // 0..1 : 32-col group
    const int lh   = lane & 15;
    const int hi   = lane >> 4;
    const int m0   = blockIdx.x * 128;
    const int n0   = blockIdx.y * 64;

    CFrag acc[2][2];
#pragma unroll
    for (int i = 0; i < 2; ++i)
#pragma unroll
        for (int j = 0; j < 2; ++j)
#pragma unroll
            for (int v = 0; v < 8; ++v) acc[i][j].e[v] = 0.f;

    const int arow = tid >> 1, ac0 = (tid & 1) * 16;   // A: 16 elems / thread
    const int brow = tid >> 3, bc0 = (tid & 7) * 8;    // B: 8 elems / thread

    for (int kb = 0; kb < D_MODEL; kb += 32) {
        // ---- stage A (128x32, f32->f16, vector LDS stores) ----
        {
            const float4* ap = reinterpret_cast<const float4*>(
                A + (size_t)(m0 + arow) * D_MODEL + kb + ac0);
            float4 r0 = ap[0], r1 = ap[1], r2 = ap[2], r3 = ap[3];
            union { f16 e[16]; uint4 q[2]; } st;
            st.e[0]=(f16)r0.x; st.e[1]=(f16)r0.y; st.e[2]=(f16)r0.z; st.e[3]=(f16)r0.w;
            st.e[4]=(f16)r1.x; st.e[5]=(f16)r1.y; st.e[6]=(f16)r1.z; st.e[7]=(f16)r1.w;
            st.e[8]=(f16)r2.x; st.e[9]=(f16)r2.y; st.e[10]=(f16)r2.z; st.e[11]=(f16)r2.w;
            st.e[12]=(f16)r3.x; st.e[13]=(f16)r3.y; st.e[14]=(f16)r3.z; st.e[15]=(f16)r3.w;
            uint4* dst = reinterpret_cast<uint4*>(&As[arow * 32 + ac0]);
            dst[0] = st.q[0]; dst[1] = st.q[1];
        }
        // ---- stage B transposed (coalesced global read, scattered LDS b16) ----
        {
            const float4* bp = reinterpret_cast<const float4*>(
                W + (size_t)(kb + brow) * D_MODEL + n0 + bc0);
            float4 r0 = bp[0], r1 = bp[1];
            const float w[8] = { r0.x, r0.y, r0.z, r0.w, r1.x, r1.y, r1.z, r1.w };
#pragma unroll
            for (int j = 0; j < 8; ++j)
                Bst[(bc0 + j) * 32 + brow] = (f16)w[j];
        }
        __syncthreads();

        // ---- fragments: contiguous b128 LDS loads ----
        AFrag a[2], b[2];
#pragma unroll
        for (int sm = 0; sm < 2; ++sm) {
            const int row = wm * 32 + sm * 16 + lh;
            a[sm].q[0] = *reinterpret_cast<const uint4*>(&As[row * 32 + hi * 8]);
            a[sm].q[1] = *reinterpret_cast<const uint4*>(&As[row * 32 + 16 + hi * 8]);
        }
#pragma unroll
        for (int sn = 0; sn < 2; ++sn) {
            const int col = wn * 32 + sn * 16 + lh;
            b[sn].q[0] = *reinterpret_cast<const uint4*>(&Bst[col * 32 + hi * 8]);
            b[sn].q[1] = *reinterpret_cast<const uint4*>(&Bst[col * 32 + 16 + hi * 8]);
        }
#pragma unroll
        for (int sm = 0; sm < 2; ++sm)
#pragma unroll
            for (int sn = 0; sn < 2; ++sn)
                acc[sm][sn].v = __builtin_amdgcn_wmma_f32_16x16x32_f16(
                    false, a[sm].v, false, b[sn].v, (short)0, acc[sm][sn].v,
                    false, false);
        __syncthreads();
    }

#pragma unroll
    for (int sn = 0; sn < 2; ++sn) {
        const int   n  = n0 + wn * 32 + sn * 16 + lh;
        const float bb = bias[n];
#pragma unroll
        for (int sm = 0; sm < 2; ++sm)
#pragma unroll
            for (int v = 0; v < 8; ++v) {
                const int   row = m0 + wm * 32 + sm * 16 + v + hi * 8;
                const float val = acc[sm][sn].e[v] + bb;
                if (F32OUT) outF[(size_t)row * D_MODEL + n] = val;
                else        outH[(size_t)row * D_MODEL + n] = (f16)val;
            }
    }
}

// ---------------------------------------------------------------------------
// Flash attention (unscaled scores, causal). One wave per 16-query tile per
// (batch, head). K/V 32x64 tiles double-buffered in LDS and staged by the
// Tensor Data Mover: block i+1's DMA overlaps block i's WMMA + softmax; the
// wave waits tensorcnt<=2 (tensor ops retire in-order) so only the current
// block is guaranteed resident. Softmax reductions via v_permlane16_b32.
// ---------------------------------------------------------------------------
__global__ __launch_bounds__(32)
void flash_attn(const f16* __restrict__ Q, const f16* __restrict__ Km,
                const f16* __restrict__ V, float* __restrict__ O)
{
    __shared__ f16 q_lds[16 * 64];
    __shared__ f16 k_lds[2][32 * 64];
    __shared__ f16 v_lds[2][32 * 64];
    __shared__ f16 p_lds[16 * 32];

    const int    lane = threadIdx.x & 31;
    const int    lh   = lane & 15;
    const int    hi   = lane >> 4;
    const int    q0   = blockIdx.x * 16;
    const int    bh   = blockIdx.y;
    const int    bb   = bh >> 4;
    const int    hh   = bh & 15;
    const size_t rowbase = (size_t)bb * SEQ;
    const int    hcol = hh * DEPTH;
    const int    kend = q0 + 15;           // last key needed (causal)

    // Q tile 16x64: each lane copies half a row (4x b128)
    {
        const int r = lane >> 1, c = (lane & 1) * 32;
        const uint4* src = reinterpret_cast<const uint4*>(
            Q + (rowbase + q0 + r) * D_MODEL + hcol + c);
        uint4* dst = reinterpret_cast<uint4*>(&q_lds[r * 64 + c]);
#pragma unroll
        for (int i = 0; i < 4; ++i) dst[i] = src[i];
    }

#if USE_TDM
    const unsigned long long gk0 =
        (unsigned long long)(uintptr_t)(Km + rowbase * D_MODEL + hcol);
    const unsigned long long gv0 =
        (unsigned long long)(uintptr_t)(V + rowbase * D_MODEL + hcol);
    const unsigned lds_k[2] = { (unsigned)(uintptr_t)(&k_lds[0][0]),
                                (unsigned)(uintptr_t)(&k_lds[1][0]) };
    const unsigned lds_v[2] = { (unsigned)(uintptr_t)(&v_lds[0][0]),
                                (unsigned)(uintptr_t)(&v_lds[1][0]) };
    // prologue: start DMA for key block 0 into buffer 0
    tdm_load_tile_32x64(gk0, lds_k[0]);
    tdm_load_tile_32x64(gv0, lds_v[0]);
#endif

    CFrag o[4];
#pragma unroll
    for (int nt = 0; nt < 4; ++nt)
#pragma unroll
        for (int v = 0; v < 8; ++v) o[nt].e[v] = 0.f;
    float mrow[8], lrow[8];
#pragma unroll
    for (int v = 0; v < 8; ++v) { mrow[v] = -3.0e38f; lrow[v] = 0.f; }

    int bufi = 0;
    for (int key0 = 0; key0 <= kend; key0 += 32, bufi ^= 1) {
#if USE_TDM
        const int keyn = key0 + 32;
        if (keyn <= kend) {
            // prefetch next block into the other buffer, overlap with compute
            tdm_load_tile_32x64(gk0 + (unsigned long long)keyn * (D_MODEL * 2),
                                lds_k[bufi ^ 1]);
            tdm_load_tile_32x64(gv0 + (unsigned long long)keyn * (D_MODEL * 2),
                                lds_v[bufi ^ 1]);
            __builtin_amdgcn_s_wait_tensorcnt(2);  // current block resident
        } else {
            __builtin_amdgcn_s_wait_tensorcnt(0);
        }
#else
        {
            const uint4* ks = reinterpret_cast<const uint4*>(
                Km + (rowbase + key0 + lane) * D_MODEL + hcol);
            const uint4* vs = reinterpret_cast<const uint4*>(
                V + (rowbase + key0 + lane) * D_MODEL + hcol);
            uint4* kd = reinterpret_cast<uint4*>(&k_lds[bufi][lane * 64]);
            uint4* vd = reinterpret_cast<uint4*>(&v_lds[bufi][lane * 64]);
#pragma unroll
            for (int i = 0; i < 8; ++i) { kd[i] = ks[i]; vd[i] = vs[i]; }
        }
#endif
        __syncthreads();

        // scores: S(16x32) = Q(16x64) . K^T  -- b128 fragment loads
        CFrag sc[2];
#pragma unroll
        for (int nt = 0; nt < 2; ++nt) {
#pragma unroll
            for (int v = 0; v < 8; ++v) sc[nt].e[v] = 0.f;
#pragma unroll
            for (int s = 0; s < 2; ++s) {
                AFrag a, bw;
                a.q[0] = *reinterpret_cast<const uint4*>(
                    &q_lds[lh * 64 + s * 32 + hi * 8]);
                a.q[1] = *reinterpret_cast<const uint4*>(
                    &q_lds[lh * 64 + s * 32 + 16 + hi * 8]);
                const f16* kp =
                    &k_lds[bufi][(nt * 16 + lh) * 64 + s * 32 + hi * 8];
                bw.q[0] = *reinterpret_cast<const uint4*>(kp);
                bw.q[1] = *reinterpret_cast<const uint4*>(kp + 16);
                sc[nt].v = __builtin_amdgcn_wmma_f32_16x16x32_f16(
                    false, a.v, false, bw.v, (short)0, sc[nt].v, false, false);
            }
        }

        // causal mask + online softmax (row r = v + 8*hi, col = lane&15)
#pragma unroll
        for (int v = 0; v < 8; ++v) {
            const int r   = v + hi * 8;
            const int qg  = q0 + r;
            const int kg0 = key0 + lh;
            float s0 = (kg0      <= qg) ? sc[0].e[v] : -1.0e9f;
            float s1 = (kg0 + 16 <= qg) ? sc[1].e[v] : -1.0e9f;
            float rm = fmaxf(s0, s1);
            rm = fmaxf(rm, XSHF16(rm, 1, SEL1_LO, SEL1_HI));
            rm = fmaxf(rm, XSHF16(rm, 2, SEL2_LO, SEL2_HI));
            rm = fmaxf(rm, XSHF16(rm, 4, SEL4_LO, SEL4_HI));
            rm = fmaxf(rm, XSHF16(rm, 8, SEL8_LO, SEL8_HI));
            const float mnew  = fmaxf(mrow[v], rm);
            const float scale = __expf(mrow[v] - mnew);
            const float p0 = __expf(s0 - mnew);
            const float p1 = __expf(s1 - mnew);
            float ps = p0 + p1;
            ps += XSHF16(ps, 1, SEL1_LO, SEL1_HI);
            ps += XSHF16(ps, 2, SEL2_LO, SEL2_HI);
            ps += XSHF16(ps, 4, SEL4_LO, SEL4_HI);
            ps += XSHF16(ps, 8, SEL8_LO, SEL8_HI);
            lrow[v] = lrow[v] * scale + ps;
            mrow[v] = mnew;
#pragma unroll
            for (int nt = 0; nt < 4; ++nt) o[nt].e[v] *= scale;
            p_lds[r * 32 + lh]      = (f16)p0;   // C-layout -> LDS
            p_lds[r * 32 + 16 + lh] = (f16)p1;
        }
        __syncthreads();

        // O(16x64) += P(16x32) . V(32x64)
        AFrag ap;
        ap.q[0] = *reinterpret_cast<const uint4*>(&p_lds[lh * 32 + hi * 16]);
        ap.q[1] = *reinterpret_cast<const uint4*>(&p_lds[lh * 32 + hi * 16 + 8]);
#pragma unroll
        for (int nt = 0; nt < 4; ++nt) {
            AFrag bw;
#pragma unroll
            for (int h = 0; h < 16; ++h)
                bw.e[h] = v_lds[bufi][(hi * 16 + h) * 64 + nt * 16 + lh];
            o[nt].v = __builtin_amdgcn_wmma_f32_16x16x32_f16(
                false, ap.v, false, bw.v, (short)0, o[nt].v, false, false);
        }
    }

#pragma unroll
    for (int nt = 0; nt < 4; ++nt)
#pragma unroll
        for (int v = 0; v < 8; ++v) {
            const int r = v + hi * 8;
            O[(rowbase + q0 + r) * D_MODEL + hcol + nt * 16 + lh] =
                o[nt].e[v] / lrow[v];
        }
}

// ---------------------------------------------------------------------------
extern "C" void kernel_launch(void* const* d_in, const int* in_sizes, int n_in,
                              void* d_out, int out_size, void* d_ws, size_t ws_size,
                              hipStream_t stream)
{
    const float* q  = (const float*)d_in[0];
    const float* k  = (const float*)d_in[1];
    const float* v  = (const float*)d_in[2];
    const float* wq = (const float*)d_in[3];
    const float* bq = (const float*)d_in[4];
    const float* wk = (const float*)d_in[5];
    const float* bk = (const float*)d_in[6];
    const float* wv = (const float*)d_in[7];
    const float* bv = (const float*)d_in[8];
    const float* wo = (const float*)d_in[9];
    const float* bo = (const float*)d_in[10];
    float* out = (float*)d_out;

    char* ws = (char*)d_ws;
    const size_t projBytes = (size_t)M_TOT * D_MODEL * sizeof(f16); // 8 MB each
    f16*   Qp = (f16*)ws;
    f16*   Kp = (f16*)(ws + projBytes);
    f16*   Vp = (f16*)(ws + 2 * projBytes);
    float* Oc = (float*)(ws + 3 * projBytes);                       // 16 MB f32

    dim3 gg(M_TOT / 128, D_MODEL / 64);
    dim3 gb(256);
    gemm_wmma<false><<<gg, gb, 0, stream>>>(q, wq, bq, Qp, nullptr);
    gemm_wmma<false><<<gg, gb, 0, stream>>>(k, wk, bk, Kp, nullptr);
    gemm_wmma<false><<<gg, gb, 0, stream>>>(v, wv, bv, Vp, nullptr);

    dim3 ag(SEQ / 16, BATCH * N_HEADS);
    flash_attn<<<ag, dim3(32), 0, stream>>>(Qp, Kp, Vp, Oc);

    gemm_wmma<true><<<gg, gb, 0, stream>>>(Oc, wo, bo, nullptr, out);
}
